// AdaptFormer_6176162972376
// MI455X (gfx1250) — matrix operand order repaired and checked by reading, not compiled, not run on hardware
//
#include <hip/hip_runtime.h>
#include <hip/hip_bf16.h>

// ---------------------------------------------------------------------------
// AdaptFormer fused block for MI455X (gfx1250, wave32, WMMA f32_16x16x32_f16)
// ---------------------------------------------------------------------------

typedef _Float16 h16_t;
typedef __attribute__((ext_vector_type(16))) _Float16 v16h;
typedef __attribute__((ext_vector_type(8)))  float    v8f;
typedef unsigned int       u32;
typedef unsigned long long u64;

union U16 { v16h v; u32 u[8]; };
union CV32 { u32 w; h16_t h[2]; };

#define NBATCH 16
#define NA 512
#define NVID 196
#define NCAT 708
#define NLAT 32
#define INV_SQRT_D 0.036084391824351615f   // 768^-0.5
#define HEAD_SCALE 0.125f                   // 64^-0.5

// A-fragment (16x32 f16): lane row = l15; dword j<4 at rp[hi*4+j], j>=4 at rp[8+hi*4+j-4]
__device__ __forceinline__ void ld_fragA(U16& f, const u32* rp, int hi)
{
    const uint4* p = (const uint4*)rp;
    uint4 a = p[hi], b = p[2 + hi];
    f.u[0] = a.x; f.u[1] = a.y; f.u[2] = a.z; f.u[3] = a.w;
    f.u[4] = b.x; f.u[5] = b.y; f.u[6] = b.z; f.u[7] = b.w;
}
// B-fragment (32x16 f16): lane col = l15; dword j<4 at rp[hi*4+j], j>=4 at rp[8+hi*4+j-4]
__device__ __forceinline__ void ld_fragB(U16& f, const u32* rp, int hi)
{
    const uint4* p = (const uint4*)rp;
    uint4 a = p[hi], b = p[2 + hi];
    f.u[0] = a.x; f.u[1] = a.y; f.u[2] = a.z; f.u[3] = a.w;
    f.u[4] = b.x; f.u[5] = b.y; f.u[6] = b.z; f.u[7] = b.w;
}

// ============================================================================
// Generic WMMA GEMM:  out[M,N] = epilogue( A16[M,K] @ op(B16) )
//   bKN==0 : B stored [N,K] row-major ("weights", computes A @ B^T)
//   bKN==1 : B stored [K,N] row-major (computes A @ B)
// epilogue: +bias[n]; act (0 none, 1 exact GELU, 2 quickGELU); *(*alpha_p);
//           +resid; store f32 (outC) and/or f16 (outH).
// Block: 64x64 tile, 128 threads (4 waves), each wave 32x32 via 2x2 WMMA.
// All leading dims are multiples of 4 halves, so b64 staging is aligned.
// ============================================================================
__global__ __launch_bounds__(128)
void wmma_gemm(const h16_t* __restrict__ A, long long sAb, int lda,
               const h16_t* __restrict__ Bw, long long sBb, int ldb, int bKN,
               float* outC, h16_t* outH, long long sOb, int ldo,
               const float* __restrict__ bias,
               const float* __restrict__ resid, long long sRb,
               const float* __restrict__ alpha_p,
               int M, int N, int K, int act)
{
    __shared__ __align__(16) h16_t sA[64 * 32];
    __shared__ __align__(16) h16_t sB[64 * 32];

    const int z = blockIdx.z;
    A  += (size_t)z * sAb;
    Bw += (size_t)z * sBb;
    if (outC)  outC  += (size_t)z * sOb;
    if (outH)  outH  += (size_t)z * sOb;
    if (resid) resid += (size_t)z * sRb;

    const int tid  = threadIdx.x;
    const int lane = tid & 31;
    const int wave = tid >> 5;
    const int wm   = wave >> 1, wn = wave & 1;
    const int l15  = lane & 15, hi = lane >> 4;
    const int m0   = blockIdx.y * 64, n0 = blockIdx.x * 64;

    const bool fullM = (m0 + 64 <= M);
    const bool fullN = (n0 + 64 <= N);

    v8f acc[2][2];
#pragma unroll
    for (int i = 0; i < 2; ++i)
#pragma unroll
        for (int j = 0; j < 2; ++j)
#pragma unroll
            for (int v = 0; v < 8; ++v) acc[i][j][v] = 0.f;

    for (int k0 = 0; k0 < K; k0 += 32) {
        __syncthreads();
        const bool fullK = (k0 + 32 <= K);
        // ---- stage A tile [64 rows x 32 halves] ----
        if (fullM && fullK) {
#pragma unroll
            for (int it = 0; it < 4; ++it) {
                int idx = tid + it * 128;        // 0..511 : row*8 + qword-chunk
                int r = idx >> 3, c = idx & 7;
                u64 v = *(const u64*)(A + (size_t)(m0 + r) * lda + k0 + c * 4);
                *(u64*)&sA[idx * 4] = v;
            }
            if (k0 + 32 < K)
                __builtin_prefetch(A + (size_t)(m0 + (tid >> 1)) * lda + k0 + 32, 0, 1);
        } else {
#pragma unroll
            for (int it = 0; it < 8; ++it) {
                int idx = tid + it * 128;        // pair index 0..1023
                int r = idx >> 4, p = idx & 15;
                int gm = m0 + r, gk = k0 + 2 * p;
                u32 val = 0u;
                if (gm < M && gk + 1 < K)        // K always even
                    val = *(const u32*)(A + (size_t)gm * lda + gk);
                *(u32*)&sA[idx * 2] = val;
            }
        }
        // ---- stage B tile as [n][k] 64x32 ----
        if (!bKN) {
            if (fullN && fullK) {
#pragma unroll
                for (int it = 0; it < 4; ++it) {
                    int idx = tid + it * 128;
                    int r = idx >> 3, c = idx & 7;
                    u64 v = *(const u64*)(Bw + (size_t)(n0 + r) * ldb + k0 + c * 4);
                    *(u64*)&sB[idx * 4] = v;
                }
                if (k0 + 32 < K)
                    __builtin_prefetch(Bw + (size_t)(n0 + (tid >> 1)) * ldb + k0 + 32, 0, 1);
            } else {
#pragma unroll
                for (int it = 0; it < 8; ++it) {
                    int idx = tid + it * 128;
                    int r = idx >> 4, p = idx & 15;
                    int gn = n0 + r, gk = k0 + 2 * p;
                    u32 val = 0u;
                    if (gn < N && gk + 1 < K)
                        val = *(const u32*)(Bw + (size_t)gn * ldb + gk);
                    *(u32*)&sB[idx * 2] = val;
                }
            }
        } else {
#pragma unroll
            for (int it = 0; it < 16; ++it) {
                int idx = tid + it * 128;        // half index 0..2047
                int n = idx >> 5, k = idx & 31;
                int gn = n0 + n, gk = k0 + k;
                h16_t v = (h16_t)0.f;
                if (gn < N && gk < K) v = Bw[(size_t)gk * ldb + gn];
                sB[n * 32 + k] = v;
            }
        }
        __syncthreads();

        // ---- fragments + WMMA ----
        U16 af[2], bf[2];
#pragma unroll
        for (int i = 0; i < 2; ++i)
            ld_fragA(af[i], (const u32*)&sA[(wm * 32 + i * 16 + l15) * 32], hi);
#pragma unroll
        for (int j = 0; j < 2; ++j)
            ld_fragB(bf[j], (const u32*)&sB[(wn * 32 + j * 16 + l15) * 32], hi);
#pragma unroll
        for (int i = 0; i < 2; ++i)
#pragma unroll
            for (int j = 0; j < 2; ++j)
                acc[i][j] = __builtin_amdgcn_wmma_f32_16x16x32_f16(
                    false, af[i].v, false, bf[j].v, (short)0, acc[i][j], false, false);
    }

    // ---- epilogue ----
    const float alpha = alpha_p ? alpha_p[0] : 1.0f;
#pragma unroll
    for (int i = 0; i < 2; ++i) {
#pragma unroll
        for (int j = 0; j < 2; ++j) {
            int gn = n0 + wn * 32 + j * 16 + l15;
            bool nok = gn < N;
            float bv = (bias && nok) ? bias[gn] : 0.f;
#pragma unroll
            for (int v = 0; v < 8; ++v) {
                int gm = m0 + wm * 32 + i * 16 + v + hi * 8;
                if (gm < M && nok) {
                    float r_ = acc[i][j][v] + bv;
                    if (act == 1)      r_ = 0.5f * r_ * (1.f + erff(r_ * 0.7071067811865476f));
                    else if (act == 2) r_ = r_ / (1.f + __expf(-1.702f * r_));
                    r_ *= alpha;
                    size_t o = (size_t)gm * ldo + gn;
                    if (resid) r_ += resid[o];
                    if (outC) outC[o] = r_;
                    if (outH) outH[o] = (h16_t)r_;
                }
            }
        }
    }
}

// ============================================================================
// Flash attention: 128 threads = 4 waves per block; each wave owns one 16-row
// q-tile; the 32-key K/V chunk is staged once in LDS and shared by all waves.
// K staged [key][dim]; V staged transposed [dim][key] so the P@V B-fragments
// are contiguous b128 LDS reads.
// qkv16: [B*Ntok, 2304] f16 (q|k|v).  o16: [B*Ntok, 768] f16 (heads concat).
// ============================================================================
__global__ __launch_bounds__(128)
void flash_attn(const h16_t* __restrict__ qkv, h16_t* __restrict__ o16, int Ntok)
{
    __shared__ __align__(16) h16_t sK[32 * 64];      // [key][dim]
    __shared__ __align__(16) h16_t sVt[64 * 32];     // [dim][key]
    __shared__ __align__(16) h16_t sP[4][16 * 32];   // per-wave P staging

    const int b = blockIdx.z, h = blockIdx.y;
    const int tid = threadIdx.x;
    const int wave = tid >> 5, lane = tid & 31;
    const int l15 = lane & 15, hi = lane >> 4;
    const int q0 = blockIdx.x * 64 + wave * 16;

    const h16_t* base = qkv + (size_t)b * Ntok * 2304;
    const h16_t* Qp = base + h * 64;
    const h16_t* Kp = base + 768 + h * 64;
    const h16_t* Vp = base + 1536 + h * 64;

    // Q fragments (b128 global reads)
    int qr = q0 + l15; if (qr >= Ntok) qr = Ntok - 1;
    const u32* qrow = (const u32*)(Qp + (size_t)qr * 2304);
    U16 qa[2];
    ld_fragA(qa[0], qrow, hi);
    ld_fragA(qa[1], qrow + 16, hi);

    float mrow[8], lrow[8];
    v8f o[4], zero;
#pragma unroll
    for (int v = 0; v < 8; ++v) { mrow[v] = -3.0e38f; lrow[v] = 0.f; zero[v] = 0.f; }
#pragma unroll
    for (int dt = 0; dt < 4; ++dt)
#pragma unroll
        for (int v = 0; v < 8; ++v) o[dt][v] = 0.f;

    for (int kc = 0; kc < Ntok; kc += 32) {
        __syncthreads();   // previous chunk fully consumed
        // ---- stage K chunk [32 x 64] (b64 loads) ----
#pragma unroll
        for (int it = 0; it < 4; ++it) {
            int idx = tid + it * 128;            // 0..511 : key*16 + qword
            int r = idx >> 4, c = idx & 15;
            int key = kc + r; if (key >= Ntok) key = Ntok - 1;
            u64 v = *(const u64*)(Kp + (size_t)key * 2304 + c * 4);
            *(u64*)&sK[idx * 4] = v;
        }
        // ---- stage V chunk transposed [64 dims x 32 keys] ----
#pragma unroll
        for (int it = 0; it < 8; ++it) {
            int idx = tid + it * 128;            // 0..1023 : key*32 + dim-pair
            int r = idx >> 5, dp = idx & 31;
            int key = kc + r; if (key >= Ntok) key = Ntok - 1;
            CV32 cv; cv.w = *(const u32*)(Vp + (size_t)key * 2304 + dp * 2);
            sVt[(dp * 2) * 32 + r]     = cv.h[0];
            sVt[(dp * 2 + 1) * 32 + r] = cv.h[1];
        }
        __syncthreads();

        // ---- S = (Q K^T) * head_scale, two 16x16 tiles ----
        v8f S[2];
#pragma unroll
        for (int t = 0; t < 2; ++t) {
            const u32* krow = (const u32*)&sK[(t * 16 + l15) * 64];
            U16 kf0, kf1;
            ld_fragA(kf0, krow, hi);
            ld_fragA(kf1, krow + 16, hi);
            v8f s = __builtin_amdgcn_wmma_f32_16x16x32_f16(
                false, qa[0].v, false, kf0.v, (short)0, zero, false, false);
            s = __builtin_amdgcn_wmma_f32_16x16x32_f16(
                false, qa[1].v, false, kf1.v, (short)0, s, false, false);
            bool oob = (kc + t * 16 + l15) >= Ntok;
#pragma unroll
            for (int v = 0; v < 8; ++v) s[v] = oob ? -3.0e38f : s[v] * HEAD_SCALE;
            S[t] = s;
        }
        // ---- online softmax (rows live in 16-lane halves of C layout) ----
        float mc[8];
#pragma unroll
        for (int v = 0; v < 8; ++v) mc[v] = fmaxf(S[0][v], S[1][v]);
        for (int msk = 1; msk < 16; msk <<= 1)
#pragma unroll
            for (int v = 0; v < 8; ++v) mc[v] = fmaxf(mc[v], __shfl_xor(mc[v], msk, 32));
        float corr[8];
#pragma unroll
        for (int v = 0; v < 8; ++v) {
            float mn = fmaxf(mrow[v], mc[v]);
            corr[v] = __expf(mrow[v] - mn);
            mrow[v] = mn;
            S[0][v] = __expf(S[0][v] - mn);
            S[1][v] = __expf(S[1][v] - mn);
        }
        float rs[8];
#pragma unroll
        for (int v = 0; v < 8; ++v) rs[v] = S[0][v] + S[1][v];
        for (int msk = 1; msk < 16; msk <<= 1)
#pragma unroll
            for (int v = 0; v < 8; ++v) rs[v] += __shfl_xor(rs[v], msk, 32);
#pragma unroll
        for (int v = 0; v < 8; ++v) lrow[v] = lrow[v] * corr[v] + rs[v];
#pragma unroll
        for (int dt = 0; dt < 4; ++dt)
#pragma unroll
            for (int v = 0; v < 8; ++v) o[dt][v] *= corr[v];

        // ---- P (C layout) -> wave-private LDS -> A-fragment ----
        // (DS ops from one wave are in-order; no block barrier needed.)
        h16_t* pw = sP[wave];
#pragma unroll
        for (int v = 0; v < 8; ++v) {
            int row = v + hi * 8;
            pw[row * 32 + l15]      = (h16_t)S[0][v];
            pw[row * 32 + 16 + l15] = (h16_t)S[1][v];
        }
        U16 pf;
        ld_fragA(pf, (const u32*)&pw[l15 * 32], hi);

        // ---- O += P @ V  (4 dim-tiles, B-fragments contiguous in sVt) ----
#pragma unroll
        for (int dt = 0; dt < 4; ++dt) {
            U16 vf;
            ld_fragB(vf, (const u32*)&sVt[(dt * 16 + l15) * 32], hi);
            o[dt] = __builtin_amdgcn_wmma_f32_16x16x32_f16(
                false, pf.v, false, vf.v, (short)0, o[dt], false, false);
        }
    }
    // ---- normalize + store f16 ----
#pragma unroll
    for (int dt = 0; dt < 4; ++dt) {
        int dim = h * 64 + dt * 16 + l15;
#pragma unroll
        for (int v = 0; v < 8; ++v) {
            int tok = q0 + v + hi * 8;
            if (tok < Ntok)
                o16[((size_t)b * Ntok + tok) * 768 + dim] = (h16_t)(o[dt][v] / lrow[v]);
        }
    }
}

// ============================================================================
// LayerNorm over 768 cols, f32 in -> f16 out.  One 256-thread block per row.
// ============================================================================
__global__ __launch_bounds__(256)
void layernorm_k(const float* __restrict__ X, const float* __restrict__ g,
                 const float* __restrict__ bb, h16_t* __restrict__ out)
{
    __shared__ float red[256];
    const size_t row = blockIdx.x;
    const float* x = X + row * 768;
    const int tid = threadIdx.x;
    float v0 = x[tid], v1 = x[tid + 256], v2 = x[tid + 512];
    red[tid] = v0 + v1 + v2; __syncthreads();
    for (int o = 128; o > 0; o >>= 1) { if (tid < o) red[tid] += red[tid + o]; __syncthreads(); }
    float mean = red[0] * (1.f / 768.f); __syncthreads();
    float d0 = v0 - mean, d1 = v1 - mean, d2 = v2 - mean;
    red[tid] = d0 * d0 + d1 * d1 + d2 * d2; __syncthreads();
    for (int o = 128; o > 0; o >>= 1) { if (tid < o) red[tid] += red[tid + o]; __syncthreads(); }
    float rstd = rsqrtf(red[0] * (1.f / 768.f) + 1e-6f);
    h16_t* orow = out + row * 768;
    orow[tid]       = (h16_t)(d0 * rstd * g[tid]       + bb[tid]);
    orow[tid + 256] = (h16_t)(d1 * rstd * g[tid + 256] + bb[tid + 256]);
    orow[tid + 512] = (h16_t)(d2 * rstd * g[tid + 512] + bb[tid + 512]);
}

// ============================================================================
// Row softmax of (logits * scale), f32 in -> f16 probs.  One block per row.
// ============================================================================
__global__ __launch_bounds__(256)
void softmax_k(const float* __restrict__ S, h16_t* __restrict__ P, int Lr, float scale)
{
    __shared__ float red[256];
    const size_t row = blockIdx.x;
    const float* s = S + row * (size_t)Lr;
    h16_t* p = P + row * (size_t)Lr;
    const int tid = threadIdx.x;
    float mx = -3.0e38f;
    for (int i = tid; i < Lr; i += 256) mx = fmaxf(mx, s[i] * scale);
    red[tid] = mx; __syncthreads();
    for (int o = 128; o > 0; o >>= 1) { if (tid < o) red[tid] = fmaxf(red[tid], red[tid + o]); __syncthreads(); }
    mx = red[0]; __syncthreads();
    float sum = 0.f;
    for (int i = tid; i < Lr; i += 256) sum += __expf(s[i] * scale - mx);
    red[tid] = sum; __syncthreads();
    for (int o = 128; o > 0; o >>= 1) { if (tid < o) red[tid] += red[tid + o]; __syncthreads(); }
    float inv = 1.f / red[0];
    for (int i = tid; i < Lr; i += 256) p[i] = (h16_t)(__expf(s[i] * scale - mx) * inv);
}

// ---- f32 -> f16 convert helpers ----
__global__ void cvt16_k(const float* __restrict__ s, h16_t* __restrict__ d, size_t n)
{
    size_t i = (size_t)blockIdx.x * blockDim.x + threadIdx.x;
    if (i < n) d[i] = (h16_t)s[i];
}

__global__ void concat16_k(const float* __restrict__ s, h16_t* __restrict__ dst,
                           int rows, int rowoff, size_t n)
{
    size_t i = (size_t)blockIdx.x * blockDim.x + threadIdx.x;
    if (i >= n) return;
    size_t per = (size_t)rows * 768;
    size_t b = i / per, rem = i - b * per;
    size_t r = rem / 768, c = rem - r * 768;
    dst[(b * NCAT + (size_t)rowoff + r) * 768 + c] = (h16_t)s[i];
}

// ============================================================================
// Launcher
// ============================================================================
extern "C" void kernel_launch(void* const* d_in, const int* in_sizes, int n_in,
                              void* d_out, int out_size, void* d_ws, size_t ws_size,
                              hipStream_t stream)
{
    (void)in_sizes; (void)n_in; (void)out_size; (void)ws_size;
    auto F = [&](int i) { return (const float*)d_in[i]; };
    const float* x       = F(0);
    const float* y       = F(1);
    const float* latents = F(2);
    const float* scale_a = F(3);
    const float* scale_v = F(4);

    // ---- workspace carve-out ----
    char* w = (char*)d_ws;
    size_t off = 0;
    auto carve = [&](size_t bytes) { void* p = w + off; off = (off + bytes + 255) & ~(size_t)255; return p; };
    h16_t* concat16 = (h16_t*)carve((size_t)NBATCH * NCAT * 768 * 2);
    h16_t* lat16    = (h16_t*)carve((size_t)NLAT * 768 * 2);
    float* s1       = (float*)carve((size_t)NBATCH * NLAT * NCAT * 4);
    h16_t* p1       = (h16_t*)carve((size_t)NBATCH * NLAT * NCAT * 2);
    h16_t* fused16  = (h16_t*)carve((size_t)NBATCH * NLAT * 768 * 2);
    float* s2       = (float*)carve((size_t)NBATCH * NA * NLAT * 4);
    h16_t* p2       = (h16_t*)carve((size_t)NBATCH * NA * NLAT * 2);
    h16_t* wqkv     = (h16_t*)carve((size_t)2304 * 768 * 2);
    h16_t* wproj    = (h16_t*)carve((size_t)768 * 768 * 2);
    h16_t* wfc1     = (h16_t*)carve((size_t)3072 * 768 * 2);
    h16_t* wfc2     = (h16_t*)carve((size_t)768 * 3072 * 2);
    h16_t* wdown    = (h16_t*)carve((size_t)64 * 768 * 2);
    h16_t* wup      = (h16_t*)carve((size_t)768 * 64 * 2);
    h16_t* hbuf     = (h16_t*)carve((size_t)NBATCH * NA * 768 * 2);
    h16_t* qkvbuf   = (h16_t*)carve((size_t)NBATCH * NA * 2304 * 2);
    h16_t* obuf     = (h16_t*)carve((size_t)NBATCH * NA * 768 * 2);
    h16_t* g1buf    = (h16_t*)carve((size_t)NBATCH * NA * 3072 * 2);
    h16_t* a1buf    = (h16_t*)carve((size_t)NBATCH * NA * 64 * 2);

    auto cvt = [&](const float* src, h16_t* dst, size_t n) {
        cvt16_k<<<dim3((unsigned)((n + 255) / 256)), dim3(256), 0, stream>>>(src, dst, n);
    };
    auto gemm = [&](const h16_t* A, long long sAb, int lda,
                    const h16_t* Bm, long long sBb, int ldb, int bKN,
                    float* oC, h16_t* oH, long long sOb, int ldo,
                    const float* bias, const float* res, long long sRb,
                    const float* alpha, int M, int N, int K, int act, int batch) {
        dim3 g((N + 63) / 64, (M + 63) / 64, batch);
        wmma_gemm<<<g, dim3(128), 0, stream>>>(A, sAb, lda, Bm, sBb, ldb, bKN,
                                               oC, oH, sOb, ldo, bias, res, sRb,
                                               alpha, M, N, K, act);
    };

    const size_t nx = (size_t)NBATCH * NA * 768;
    const size_t ny = (size_t)NBATCH * NVID * 768;
    float* Tx = (float*)d_out;
    float* Ty = Tx + nx;

    // ================= fusion via latent bottleneck =================
    cvt(latents, lat16, (size_t)NLAT * 768);
    concat16_k<<<dim3((unsigned)((nx + 255) / 256)), dim3(256), 0, stream>>>(x, concat16, NA, 0, nx);
    concat16_k<<<dim3((unsigned)((ny + 255) / 256)), dim3(256), 0, stream>>>(y, concat16, NVID, NA, ny);

    // s1[b,32,708] = lat @ concat^T
    gemm(lat16, 0, 768, concat16, (long long)NCAT * 768, 768, 0,
         s1, nullptr, (long long)NLAT * NCAT, NCAT,
         nullptr, nullptr, 0, nullptr, NLAT, NCAT, 768, 0, NBATCH);
    softmax_k<<<dim3(NBATCH * NLAT), dim3(256), 0, stream>>>(s1, p1, NCAT, INV_SQRT_D);
    // fused[b,32,768] = p1 @ concat
    gemm(p1, (long long)NLAT * NCAT, NCAT, concat16, (long long)NCAT * 768, 768, 1,
         nullptr, fused16, (long long)NLAT * 768, 768,
         nullptr, nullptr, 0, nullptr, NLAT, 768, NCAT, 0, NBATCH);

    // x-stream:  Tx = x + scale_a * softmax(x fused^T * D^-0.5) fused
    gemm(concat16, (long long)NCAT * 768, 768, fused16, (long long)NLAT * 768, 768, 0,
         s2, nullptr, (long long)NA * NLAT, NLAT,
         nullptr, nullptr, 0, nullptr, NA, NLAT, 768, 0, NBATCH);
    softmax_k<<<dim3(NBATCH * NA), dim3(256), 0, stream>>>(s2, p2, NLAT, INV_SQRT_D);
    gemm(p2, (long long)NA * NLAT, NLAT, fused16, (long long)NLAT * 768, 768, 1,
         Tx, nullptr, (long long)NA * 768, 768,
         nullptr, x, (long long)NA * 768, scale_a, NA, 768, NLAT, 0, NBATCH);

    // y-stream (reuses s2/p2)
    gemm(concat16 + (size_t)NA * 768, (long long)NCAT * 768, 768,
         fused16, (long long)NLAT * 768, 768, 0,
         s2, nullptr, (long long)NVID * NLAT, NLAT,
         nullptr, nullptr, 0, nullptr, NVID, NLAT, 768, 0, NBATCH);
    softmax_k<<<dim3(NBATCH * NVID), dim3(256), 0, stream>>>(s2, p2, NLAT, INV_SQRT_D);
    gemm(p2, (long long)NVID * NLAT, NLAT, fused16, (long long)NLAT * 768, 768, 1,
         Ty, nullptr, (long long)NVID * 768, 768,
         nullptr, y, (long long)NVID * 768, scale_v, NVID, 768, NLAT, 0, NBATCH);

    // ================= per-stream ViT block + adapter =================
    auto run_stream = [&](float* T, int Ntok, int base) {
        const float* ln1g  = F(base + 0);  const float* ln1b = F(base + 1);
        const float* qkvw  = F(base + 2);  const float* qkvb = F(base + 3);
        const float* projw = F(base + 4);  const float* projb = F(base + 5);
        const float* ln2g  = F(base + 6);  const float* ln2b = F(base + 7);
        const float* fc1w  = F(base + 8);  const float* fc1b = F(base + 9);
        const float* fc2w  = F(base + 10); const float* fc2b = F(base + 11);
        const float* downw = F(base + 12); const float* downb = F(base + 13);
        const float* upw   = F(base + 14); const float* upb  = F(base + 15);
        const float* scp   = F(base + 16);
        const int M = NBATCH * Ntok;

        cvt(qkvw, wqkv, (size_t)2304 * 768);
        cvt(projw, wproj, (size_t)768 * 768);
        cvt(fc1w, wfc1, (size_t)3072 * 768);
        cvt(fc2w, wfc2, (size_t)768 * 3072);
        cvt(downw, wdown, (size_t)64 * 768);
        cvt(upw, wup, (size_t)768 * 64);

        // h = LN1(T)
        layernorm_k<<<dim3(M), dim3(256), 0, stream>>>(T, ln1g, ln1b, hbuf);
        // qkv = h @ qkv_w^T + b  (f16 out, [M,2304])
        gemm(hbuf, 0, 768, wqkv, 0, 768, 0, nullptr, qkvbuf, 0, 2304,
             qkvb, nullptr, 0, nullptr, M, 2304, 768, 0, 1);
        // MHSA (flash, 4 waves/block share K/V chunk via LDS)
        flash_attn<<<dim3((Ntok + 63) / 64, 12, NBATCH), dim3(128), 0, stream>>>(qkvbuf, obuf, Ntok);
        // T = T + o @ proj^T + b  (in-place residual)
        gemm(obuf, 0, 768, wproj, 0, 768, 0, T, nullptr, 0, 768,
             projb, T, 0, nullptr, M, 768, 768, 0, 1);
        // h = LN2(T)
        layernorm_k<<<dim3(M), dim3(256), 0, stream>>>(T, ln2g, ln2b, hbuf);
        // mlp hidden = GELU(h @ fc1^T + b)
        gemm(hbuf, 0, 768, wfc1, 0, 768, 0, nullptr, g1buf, 0, 3072,
             fc1b, nullptr, 0, nullptr, M, 3072, 768, 1, 1);
        // adapter hidden = quickGELU(h @ down^T + b)
        gemm(hbuf, 0, 768, wdown, 0, 768, 0, nullptr, a1buf, 0, 64,
             downb, nullptr, 0, nullptr, M, 64, 768, 2, 1);
        // T += mlp_hidden @ fc2^T + b
        gemm(g1buf, 0, 3072, wfc2, 0, 3072, 0, T, nullptr, 0, 768,
             fc2b, T, 0, nullptr, M, 768, 3072, 0, 1);
        // T += scale * (adapter_hidden @ up^T + b)
        gemm(a1buf, 0, 64, wup, 0, 64, 0, T, nullptr, 0, 768,
             upb, T, 0, scp, M, 768, 64, 0, 1);
    };

    run_stream(Tx, NA, 5);     // spec stream
    run_stream(Ty, NVID, 22);  // rgb stream
}
